// Force_35502199669492
// MI455X (gfx1250) — matrix operand order repaired and checked by reading, not compiled, not run on hardware
//
#include <hip/hip_runtime.h>
#include <hip/hip_bf16.h>
#include <math.h>

typedef __attribute__((ext_vector_type(16))) __bf16 v16bf;
typedef __attribute__((ext_vector_type(8)))  __bf16 v8bf;
typedef __attribute__((ext_vector_type(8)))  float  v8f;

#define F 64
#define ROWS_PER_BLOCK 128   // 8 waves * 16 rows
#define EPS 1e-5f

__device__ __forceinline__ float softplusf(float x) {
    return fmaxf(x, 0.0f) + log1pf(__expf(-fabsf(x)));
}

// -------------------------------------------------------------------------
// Layer kernel: out[E,64](bf16) = act(in)[E,64] @ W[64,64] + b, where act is
// (optionally) batchnorm(mu,rstd,g,be) + softplus applied to `in`.
// IN_BF16: input activation buffer is bf16 (layers 1,2); else f32 (layer 0).
// Accumulates per-feature sum / sum-of-squares of the *stored (rounded)*
// output into statsOut[0..63] / statsOut[64..127] so BN stats exactly match
// the bf16 data the next layer reads.
// bf16 WMMA, f32 accumulate. In-place (in == out) is safe: each wave reads
// its 16 rows fully into registers before storing.
// -------------------------------------------------------------------------
template <bool IN_BF16, bool APPLY_BN, typename InT>
__global__ void __launch_bounds__(256)
layer_wmma(const InT* __restrict__ in, __bf16* __restrict__ out,
           const float* __restrict__ w,  const float* __restrict__ bias,
           float* __restrict__ statsOut,
           const float* __restrict__ murstd,   // [0..63]=mu, [64..127]=rstd
           const float* __restrict__ g, const float* __restrict__ be)
{
    // B operand staging: 8 combos = 4 n-tiles x 2 k-halves, each 32 lanes x 16 bf16
    __shared__ __align__(32) __bf16 Bst[8 * 32 * 16];
    __shared__ float ldsStats[128];

    const int tid  = threadIdx.x;
    const int lane = tid & 31;
    const int wv   = tid >> 5;
    const int half = lane >> 4;   // 0 or 1
    const int m    = lane & 15;

    if (tid < 128) ldsStats[tid] = 0.0f;

    // ---- stage W (64x64 f32) -> bf16 in per-lane B layout -----------------
    // B fragment layout (16x16x32 bf16): lane L holds column n = nt*16 + (L&15),
// elements j=0..7 map to K = kh*32 + (L>>4)*8 + j; j=8..15 -> K = kh*32 + (L>>4)*8 + 8 + j.
    #pragma unroll
    for (int t2 = 0; t2 < 16; ++t2) {
        int idx   = tid * 16 + t2;        // 0..4095
        int combo = idx >> 9;             // 0..7 : nt*2 + kh
        int rem   = idx & 511;
        int L     = rem >> 4;             // 0..31
        int j     = rem & 15;             // 0..15
        int nt    = combo >> 1;
        int kh    = combo & 1;
        int c     = nt * 16 + (L & 15);
        int k     = kh * 32 + ((L >> 4) << 3) + j + ((j >= 8) ? 8 : 0);
        Bst[combo * 512 + L * 16 + j] = (__bf16)w[k * F + c];
    }
    __syncthreads();

    // ---- load A fragments (both K-halves) with optional BN+softplus -------
    // A layout (16x32 bf16): lane L holds row M = L&15; for k-half kh,
    // j=0..7 -> K = kh*32 + half*8 + j ; j=8..15 -> K = kh*32 + half*8 + 8 + j.
    const long long row = (long long)blockIdx.x * ROWS_PER_BLOCK + wv * 16 + m;
    const InT* inRow    = in + row * F;

    v16bf a[2];
    #pragma unroll
    for (int kh = 0; kh < 2; ++kh) {
        int kbase = kh * 32 + half * 8;
        float buf[16];
        if (IN_BF16) {
            v8bf c0 = *(const v8bf*)((const __bf16*)inRow + kbase);
            v8bf c1 = *(const v8bf*)((const __bf16*)inRow + kbase + 16);
            #pragma unroll
            for (int j = 0; j < 8; ++j) { buf[j] = (float)c0[j]; buf[8 + j] = (float)c1[j]; }
        } else {
            const float* ir = (const float*)inRow;
            float4 c0a = *(const float4*)(ir + kbase);
            float4 c0b = *(const float4*)(ir + kbase + 4);
            float4 c1a = *(const float4*)(ir + kbase + 16);
            float4 c1b = *(const float4*)(ir + kbase + 20);
            buf[0]=c0a.x; buf[1]=c0a.y; buf[2]=c0a.z; buf[3]=c0a.w;
            buf[4]=c0b.x; buf[5]=c0b.y; buf[6]=c0b.z; buf[7]=c0b.w;
            buf[8]=c1a.x; buf[9]=c1a.y; buf[10]=c1a.z; buf[11]=c1a.w;
            buf[12]=c1b.x; buf[13]=c1b.y; buf[14]=c1b.z; buf[15]=c1b.w;
        }
        #pragma unroll
        for (int j = 0; j < 16; ++j) {
            float v = buf[j];
            if (APPLY_BN) {
                int k = kbase + j + ((j >= 8) ? 8 : 0);
                v = (v - murstd[k]) * murstd[64 + k] * g[k] + be[k];
                v = softplusf(v);
            }
            a[kh][j] = (__bf16)v;
        }
    }

    // ---- 4 output tiles of 16x16, K=64 via two 16x16x32 WMMAs ------------
    __bf16* outBase = out + ((long long)blockIdx.x * ROWS_PER_BLOCK + wv * 16 + half * 8) * F;
    #pragma unroll
    for (int nt = 0; nt < 4; ++nt) {
        v16bf b0 = *(const v16bf*)&Bst[(nt * 2 + 0) * 512 + lane * 16];
        v16bf b1 = *(const v16bf*)&Bst[(nt * 2 + 1) * 512 + lane * 16];
        v8f c = {};
        c = __builtin_amdgcn_wmma_f32_16x16x32_bf16(false, a[0], false, b0,
                                                    (short)0, c, false, false);
        c = __builtin_amdgcn_wmma_f32_16x16x32_bf16(false, a[1], false, b1,
                                                    (short)0, c, false, false);
        // C/D layout: lane holds column n = nt*16 + (lane&15); VGPR j -> row
        // (half*8 + j) of this wave's strip.
        int n    = nt * 16 + m;
        float bv = bias[n];
        float s = 0.0f, s2 = 0.0f;
        #pragma unroll
        for (int j = 0; j < 8; ++j) {
            __bf16 hb = (__bf16)(c[j] + bv);      // round first ...
            outBase[(long long)j * F + n] = hb;
            float v = (float)hb;                  // ... stats match stored data
            s  += v;
            s2 += v * v;
        }
        atomicAdd(&ldsStats[n],      s);
        atomicAdd(&ldsStats[64 + n], s2);
    }

    __syncthreads();
    if (tid < 128) atomicAdd(&statsOut[tid], ldsStats[tid]);
}

// -------------------------------------------------------------------------
__global__ void finalize_stats(const float* __restrict__ sums,
                               float* __restrict__ murstd, float invE)
{
    int n = threadIdx.x;   // 64 threads
    float mu  = sums[n] * invE;
    float var = sums[64 + n] * invE - mu * mu;
    murstd[n]      = mu;
    murstd[64 + n] = rsqrtf(var + EPS);
}

// -------------------------------------------------------------------------
__global__ void zero_kernel(float* __restrict__ out, int n,
                            float* __restrict__ stats, int ns)
{
    int i = blockIdx.x * blockDim.x + threadIdx.x;
    if (i < n)  out[i]  = 0.0f;
    if (i < ns) stats[i] = 0.0f;
}

// -------------------------------------------------------------------------
// Final pass: per-edge BN2+softplus, 64->1 dot, direction normalization,
// atomic segment-sum into out[num_nodes,3].
// -------------------------------------------------------------------------
__global__ void __launch_bounds__(256)
force_kernel(const __bf16* __restrict__ h,
             const float* __restrict__ murstd, const float* __restrict__ g,
             const float* __restrict__ be,
             const float* __restrict__ w_out, const float* __restrict__ b_out,
             const float* __restrict__ pos, const float* __restrict__ shift,
             const long long* __restrict__ eidx,
             float* __restrict__ fout, int E)
{
    int e = blockIdx.x * blockDim.x + threadIdx.x;
    if (e >= E) return;

    const v8bf* hr = (const v8bf*)(h + (long long)e * F);
    float acc = b_out[0];
    #pragma unroll
    for (int c8 = 0; c8 < 8; ++c8) {
        v8bf v8 = hr[c8];
        #pragma unroll
        for (int u = 0; u < 8; ++u) {
            int k   = c8 * 8 + u;
            float t = ((float)v8[u] - murstd[k]) * murstd[64 + k] * g[k] + be[k];
            t = softplusf(t);
            acc += t * w_out[k];
        }
    }

    long long jn  = eidx[e];
    long long in_ = eidx[E + e];
    float dx = pos[in_ * 3 + 0] + shift[(long long)e * 3 + 0] - pos[jn * 3 + 0];
    float dy = pos[in_ * 3 + 1] + shift[(long long)e * 3 + 1] - pos[jn * 3 + 1];
    float dz = pos[in_ * 3 + 2] + shift[(long long)e * 3 + 2] - pos[jn * 3 + 2];
    float s = acc * __frsqrt_rn(dx * dx + dy * dy + dz * dz);

    atomicAdd(&fout[in_ * 3 + 0], s * dx);
    atomicAdd(&fout[in_ * 3 + 1], s * dy);
    atomicAdd(&fout[in_ * 3 + 2], s * dz);
}

// -------------------------------------------------------------------------
extern "C" void kernel_launch(void* const* d_in, const int* in_sizes, int n_in,
                              void* d_out, int out_size, void* d_ws, size_t ws_size,
                              hipStream_t stream)
{
    const float*     pos    = (const float*)d_in[0];
    const float*     shift  = (const float*)d_in[1];
    const float*     ea     = (const float*)d_in[2];
    const long long* eidx   = (const long long*)d_in[3];
    const float* w0  = (const float*)d_in[4];
    const float* b0  = (const float*)d_in[5];
    const float* g0  = (const float*)d_in[6];
    const float* be0 = (const float*)d_in[7];
    const float* w1  = (const float*)d_in[8];
    const float* b1  = (const float*)d_in[9];
    const float* g1  = (const float*)d_in[10];
    const float* be1 = (const float*)d_in[11];
    const float* w2  = (const float*)d_in[12];
    const float* b2  = (const float*)d_in[13];
    const float* g2  = (const float*)d_in[14];
    const float* be2 = (const float*)d_in[15];
    const float* w_out = (const float*)d_in[16];
    const float* b_out = (const float*)d_in[17];

    const int E = in_sizes[2] / F;          // 1,600,000 (divisible by 128)

    // workspace layout: h [E*64 bf16] | sums [3*128 f32] | murstd [3*128 f32]
    __bf16* h     = (__bf16*)d_ws;
    float* sums   = (float*)((char*)d_ws + (size_t)E * F * sizeof(__bf16));
    float* murstd = sums + 3 * 128;

    // zero output + stats accumulators
    {
        int nz = (out_size > 3 * 128) ? out_size : 3 * 128;
        zero_kernel<<<(nz + 255) / 256, 256, 0, stream>>>(
            (float*)d_out, out_size, sums, 3 * 128);
    }

    const int grid = E / ROWS_PER_BLOCK;
    const float invE = 1.0f / (float)E;

    // layer 0: raw edge_attr (f32) -> h (bf16), stats0
    layer_wmma<false, false><<<grid, 256, 0, stream>>>(
        ea, h, w0, b0, sums + 0, (const float*)nullptr,
        (const float*)nullptr, (const float*)nullptr);
    finalize_stats<<<1, 64, 0, stream>>>(sums + 0, murstd + 0, invE);

    // layer 1: bn0+softplus(h) @ w1 -> h (in-place), stats1
    layer_wmma<true, true><<<grid, 256, 0, stream>>>(
        (const __bf16*)h, h, w1, b1, sums + 128, murstd + 0, g0, be0);
    finalize_stats<<<1, 64, 0, stream>>>(sums + 128, murstd + 128, invE);

    // layer 2: bn1+softplus(h) @ w2 -> h (in-place), stats2
    layer_wmma<true, true><<<grid, 256, 0, stream>>>(
        (const __bf16*)h, h, w2, b2, sums + 256, murstd + 128, g1, be1);
    finalize_stats<<<1, 64, 0, stream>>>(sums + 256, murstd + 256, invE);

    // final: bn2+softplus, 64->1 projection, force scatter
    force_kernel<<<(E + 255) / 256, 256, 0, stream>>>(
        h, murstd + 256, g2, be2, w_out, b_out, pos, shift, eidx,
        (float*)d_out, E);
}
